// Attention_33002528702591
// MI455X (gfx1250) — matrix operand order repaired and checked by reading, not compiled, not run on hardware
//
#include <hip/hip_runtime.h>
#include <hip/hip_bf16.h>
#include <stdint.h>

#define DIMD    2048
#define NHEADS  16
#define HEADDIM 128
#define SEQL    2048
#define BATCH   2
#define MROWS   (BATCH * SEQL) /* 4096 */

typedef __attribute__((ext_vector_type(16))) __bf16 v16bf;
typedef __attribute__((ext_vector_type(8)))  float  v8f;
typedef __attribute__((ext_vector_type(4)))  int    i32x4;

// ---------- CDNA5 async global->LDS path (guarded; falls back to vmem+ds) ----
#if __has_builtin(__builtin_amdgcn_global_load_async_to_lds_b128) && \
    __has_builtin(__builtin_amdgcn_s_wait_asynccnt)
#define USE_ASYNC_LDS 1
#endif

__device__ __forceinline__ void copy16_g2l(void* lds, const void* g) {
#ifdef USE_ASYNC_LDS
  // (global v4i*, local v4i*, imm offset, imm cpol)
  __builtin_amdgcn_global_load_async_to_lds_b128(
      (__attribute__((address_space(1))) i32x4*)g,
      (__attribute__((address_space(3))) i32x4*)lds,
      0, 0);
#else
  *(uint4*)lds = *(const uint4*)g;
#endif
}
__device__ __forceinline__ void copy_join() {
#ifdef USE_ASYNC_LDS
  __builtin_amdgcn_s_wait_asynccnt(0);
#endif
}

// ---------- scalar bf16 helpers ------------------------------------------------
__device__ __forceinline__ unsigned short f2bfu(float f) {
  unsigned u = __builtin_bit_cast(unsigned, f);
  u += 0x7FFFu + ((u >> 16) & 1u);           // round-to-nearest-even
  return (unsigned short)(u >> 16);
}
__device__ __forceinline__ float bf2f(unsigned short u) {
  unsigned v = ((unsigned)u) << 16;
  return __builtin_bit_cast(float, v);
}

// ---------- WMMA fragment builders (contiguous k order, k = 16*half + s) -------------
union FragU { v16bf v; unsigned u[8]; unsigned short s[16]; };

// A-matrix 16x32 bf16: lane m<16 holds row m with K=0..15 (half 0) or
// K=16..31 (half 1) -> 8 contiguous dwords at dword offset 8*half.
__device__ __forceinline__ v16bf frag_a_row(const unsigned short* row, unsigned half) {
  const unsigned* r = (const unsigned*)row;
  FragU f;
#pragma unroll
  for (int i = 0; i < 4; ++i) {
    f.u[i]     = r[half * 8 + i];
    f.u[4 + i] = r[half * 8 + 4 + i];
  }
  return f.v;
}

// B-matrix 32x16 bf16: lane n<16 holds col n with K=0..15 (half 0) or
// K=16..31 (half 1) -> 8 contiguous dwords at dword offset 8*half.
__device__ __forceinline__ v16bf frag_b_row(const unsigned short* row, unsigned half) {
  const unsigned* r = (const unsigned*)row;
  FragU f;
#pragma unroll
  for (int i = 0; i < 8; ++i) f.u[i] = r[half * 8 + i];
  return f.v;
}

// B-matrix column gather (strided): lane holds col; element k = colBase[k*stride],
// k = 16*half + 2r, 2r+1.  maxk clamps row index (causal tail OOB guard).
__device__ __forceinline__ v16bf frag_b_col(const unsigned short* col, int stride,
                                            unsigned half, int maxk) {
  FragU f;
#pragma unroll
  for (int r = 0; r < 8; ++r) {
    int k0 = (int)half * 16 + 2 * r;
    int ka = k0 < maxk ? k0 : maxk;
    int kb = (k0 + 1) < maxk ? (k0 + 1) : maxk;
    f.s[2 * r]     = col[(size_t)ka * stride];
    f.s[2 * r + 1] = col[(size_t)kb * stride];
  }
  return f.v;
}

__device__ __forceinline__ v8f wmma_bf16(v16bf a, v16bf b, v8f c) {
  return __builtin_amdgcn_wmma_f32_16x16x32_bf16(
      /*neg_a=*/false, a, /*neg_b=*/false, b,
      /*c_mod=*/(short)0, c, /*reuse_a=*/false, /*reuse_b=*/false);
}
__device__ __forceinline__ v8f vzero8() {
  v8f z = {0.f, 0.f, 0.f, 0.f, 0.f, 0.f, 0.f, 0.f};
  return z;
}

// ---------- kernel 1: fp32 -> bf16 convert ------------------------------------
__global__ __launch_bounds__(256) void cvt_f32_bf16(const float* __restrict__ in,
                                                    unsigned short* __restrict__ out,
                                                    int n) {
  int i = blockIdx.x * 256 + threadIdx.x;
  if (i < n) out[i] = f2bfu(in[i]);
}

// ---------- kernel 2: C[M,N] = A[M,K] x B[N,K]^T  (bf16 in, bf16/f32 out) -----
// 128x128 block tile, 32-wide K tiles, double-buffered LDS with async copies
// overlapping the WMMA chain.  8 waves: wave w owns rows w*16..w*16+15, all
// 128 columns -> 8 accumulators, 8 WMMAs per K step.
template <int OUT_F32>
__global__ __launch_bounds__(256) void gemm_bt(const unsigned short* __restrict__ A,
                                               const unsigned short* __restrict__ B,
                                               void* __restrict__ Cout,
                                               int M, int N, int K) {
  __shared__ unsigned short aT[2][128 * 32];  // 2 x 8 KB
  __shared__ unsigned short bT[2][128 * 32];  // 2 x 8 KB
  const int tid  = threadIdx.x;
  const int lane = tid & 31;
  const int wv   = tid >> 5;
  const int ln   = lane & 15;
  const unsigned half = (lane >> 4) & 1u;
  const int m0 = blockIdx.x * 128;
  const int n0 = blockIdx.y * 128;

  v8f acc[8];
#pragma unroll
  for (int t = 0; t < 8; ++t) acc[t] = vzero8();

  // stage one 128x32 A tile + 128x32 B tile (each 512 x 16B chunks, 2/thread)
  auto stage = [&](int buf, int k0) {
#pragma unroll
    for (int c = 0; c < 2; ++c) {
      int i = tid + c * 256;
      int row = i >> 2, col = i & 3;
      copy16_g2l(&aT[buf][row * 32 + col * 8],
                 &A[(size_t)(m0 + row) * K + k0 + col * 8]);
      copy16_g2l(&bT[buf][row * 32 + col * 8],
                 &B[(size_t)(n0 + row) * K + k0 + col * 8]);
    }
  };

  const int nk = K / 32;
  stage(0, 0);                                  // prologue
  for (int i = 0; i < nk; ++i) {
    const int cur = i & 1;
    copy_join();                                // tile i landed in LDS
    __syncthreads();
    if (i + 1 < nk) stage(1 - cur, (i + 1) * 32);  // overlap with compute

    v16bf af = frag_a_row(&aT[cur][(wv * 16 + ln) * 32], half);
#pragma unroll
    for (int t = 0; t < 8; ++t) {
      v16bf bf_ = frag_b_row(&bT[cur][(t * 16 + ln) * 32], half);
      acc[t] = wmma_bf16(af, bf_, acc[t]);
    }
    __syncthreads();                            // done reading buf[cur]
  }

  // D layout: VGPR r -> row r (lanes 0-15) / row r+8 (lanes 16-31), col = ln
#pragma unroll
  for (int t = 0; t < 8; ++t)
#pragma unroll
    for (int r = 0; r < 8; ++r) {
      int row = m0 + wv * 16 + r + (int)half * 8;
      int col = n0 + t * 16 + ln;
      float v = acc[t][r];
      if (OUT_F32) ((float*)Cout)[(size_t)row * N + col] = v;
      else         ((unsigned short*)Cout)[(size_t)row * N + col] = f2bfu(v);
    }
}

// ---------- kernel 3: RoPE on Q and K (bf16 in place) -------------------------
__global__ __launch_bounds__(256) void rope_qk(unsigned short* __restrict__ Q,
                                               unsigned short* __restrict__ Kb,
                                               const float* __restrict__ cosT,
                                               const float* __restrict__ sinT,
                                               int total) {
  int p = blockIdx.x * 256 + threadIdx.x;
  if (p >= total) return;
  int row = p >> 10;            // H*64 = 1024 pairs per row
  int rem = p & 1023;
  int h = rem >> 6, j = rem & 63;
  int s = row & (SEQL - 1);     // row = b*SEQ + s, SEQ power of 2
  float c  = cosT[s * 64 + j];
  float sn = sinT[s * 64 + j];
  size_t idx = (size_t)row * DIMD + h * HEADDIM + 2 * j;

  float qe = bf2f(Q[idx]), qo = bf2f(Q[idx + 1]);
  Q[idx]     = f2bfu(qe * c - qo * sn);
  Q[idx + 1] = f2bfu(qe * sn + qo * c);
  float ke = bf2f(Kb[idx]), ko = bf2f(Kb[idx + 1]);
  Kb[idx]     = f2bfu(ke * c - ko * sn);
  Kb[idx + 1] = f2bfu(ke * sn + ko * c);
}

// ---------- kernel 4: causal flash attention ----------------------------------
// grid (S/128, H, B); 8 waves/block; wave owns 16 query rows.
__global__ __launch_bounds__(256) void flash_attn(const unsigned short* __restrict__ Q,
                                                  const unsigned short* __restrict__ Kb,
                                                  const unsigned short* __restrict__ V,
                                                  unsigned short* __restrict__ O) {
  __shared__ unsigned short pT[8][16 * 32];   // per-wave P staging (1 KB each)
  const int lane = threadIdx.x & 31;
  const int wv   = threadIdx.x >> 5;
  const int ln   = lane & 15;
  const unsigned half = (lane >> 4) & 1u;
  const int h = blockIdx.y, b = blockIdx.z;
  const int qbase = blockIdx.x * 128 + wv * 16;
  const float scale = 0.08838834764831845f;   // 1/sqrt(128)

  // Q as 4 A-fragments (K = 128 = 4 x 32)
  v16bf qf[4];
  const unsigned short* qrow =
      Q + ((size_t)(b * SEQL + qbase + ln)) * DIMD + h * HEADDIM;
#pragma unroll
  for (int c = 0; c < 4; ++c) qf[c] = frag_a_row(qrow + c * 32, half);

  v8f o[8];
#pragma unroll
  for (int t = 0; t < 8; ++t) o[t] = vzero8();
  float mrow[8], lrow[8];
#pragma unroll
  for (int r = 0; r < 8; ++r) { mrow[r] = -3.0e38f; lrow[r] = 0.f; }

  const int kend = qbase + 16;                // exclusive causal key bound
  for (int kb = 0; kb < kend; kb += 32) {
    // ---- preload all K fragments for both 16-key tiles ------------------
    int r0 = kb + ln;
    int r1 = kb + 16 + ln; if (r1 > SEQL - 1) r1 = SEQL - 1;   // OOB clamp
    const unsigned short* krow0 =
        Kb + ((size_t)(b * SEQL + r0)) * DIMD + h * HEADDIM;
    const unsigned short* krow1 =
        Kb + ((size_t)(b * SEQL + r1)) * DIMD + h * HEADDIM;
    v16bf k0f[4], k1f[4];
#pragma unroll
    for (int c = 0; c < 4; ++c) {
      k0f[c] = frag_b_row(krow0 + c * 32, half);
      k1f[c] = frag_b_row(krow1 + c * 32, half);
    }

    // ---- scores: 8 WMMAs, K-dim 128 -------------------------------------
    v8f s0 = vzero8(), s1 = vzero8();
#pragma unroll
    for (int c = 0; c < 4; ++c) s0 = wmma_bf16(qf[c], k0f[c], s0);
#pragma unroll
    for (int c = 0; c < 4; ++c) s1 = wmma_bf16(qf[c], k1f[c], s1);

    // ---- causal mask + scale (D layout: row = r + 8*half, col = ln) -----
    float rmax[8];
#pragma unroll
    for (int r = 0; r < 8; ++r) {
      int qg = qbase + r + (int)half * 8;
      float v0 = (kb + ln      <= qg) ? s0[r] * scale : -3.0e38f;
      float v1 = (kb + 16 + ln <= qg) ? s1[r] * scale : -3.0e38f;
      s0[r] = v0; s1[r] = v1;
      rmax[r] = fmaxf(v0, v1);
    }
    // row max across the 16 lanes of each half (xor-shuffles stay in-group)
#pragma unroll
    for (int d = 1; d < 16; d <<= 1)
#pragma unroll
      for (int r = 0; r < 8; ++r)
        rmax[r] = fmaxf(rmax[r], __shfl_xor(rmax[r], d, 32));

    // ---- online softmax update -----------------------------------------
    float rsum[8];
#pragma unroll
    for (int r = 0; r < 8; ++r) {
      float mnew  = fmaxf(mrow[r], rmax[r]);
      float alpha = __expf(mrow[r] - mnew);
      mrow[r] = mnew;
      float p0 = (s0[r] < -1.0e37f) ? 0.f : __expf(s0[r] - mnew);
      float p1 = (s1[r] < -1.0e37f) ? 0.f : __expf(s1[r] - mnew);
      s0[r] = p0; s1[r] = p1;
      rsum[r] = p0 + p1;
      lrow[r] *= alpha;
#pragma unroll
      for (int t = 0; t < 8; ++t) o[t][r] *= alpha;
    }
#pragma unroll
    for (int d = 1; d < 16; d <<= 1)
#pragma unroll
      for (int r = 0; r < 8; ++r)
        rsum[r] += __shfl_xor(rsum[r], d, 32);
#pragma unroll
    for (int r = 0; r < 8; ++r) lrow[r] += rsum[r];

    // ---- P (16x32) through LDS into A-fragment layout -------------------
    unsigned short* pw = pT[wv];
#pragma unroll
    for (int r = 0; r < 8; ++r) {
      int row = r + (int)half * 8;
      pw[row * 32 + ln]      = f2bfu(s0[r]);
      pw[row * 32 + ln + 16] = f2bfu(s1[r]);
    }
    asm volatile("s_wait_dscnt 0x0" ::: "memory");   // per-wave LDS RAW fence
    v16bf pf = frag_a_row(&pw[ln * 32], half);

    // ---- O += P x V : 8 head-dim tiles ----------------------------------
    int maxk = (SEQL - 1) - kb;
#pragma unroll
    for (int t = 0; t < 8; ++t) {
      const unsigned short* vcol =
          V + ((size_t)(b * SEQL + kb)) * DIMD + h * HEADDIM + t * 16 + ln;
      v16bf vf = frag_b_col(vcol, DIMD, half, maxk);
      o[t] = wmma_bf16(pf, vf, o[t]);
    }
  }

  // ---- normalize + store (attn layout: [b*S+s, h*128+d] bf16) -------------
#pragma unroll
  for (int r = 0; r < 8; ++r) {
    float inv = 1.0f / lrow[r];
    int row = b * SEQL + qbase + r + (int)half * 8;
#pragma unroll
    for (int t = 0; t < 8; ++t) {
      int col = h * HEADDIM + t * 16 + ln;
      O[(size_t)row * DIMD + col] = f2bfu(o[t][r] * inv);
    }
  }
}

// ---------- host-side orchestration -------------------------------------------
extern "C" void kernel_launch(void* const* d_in, const int* in_sizes, int n_in,
                              void* d_out, int out_size, void* d_ws, size_t ws_size,
                              hipStream_t stream) {
  const float* x    = (const float*)d_in[0];
  // d_in[1] = start_pos (0), d_in[4] = mask: causal handled in-kernel
  const float* fcos = (const float*)d_in[2];
  const float* fsin = (const float*)d_in[3];
  const float* wq   = (const float*)d_in[5];
  const float* wk   = (const float*)d_in[6];
  const float* wv   = (const float*)d_in[7];
  const float* wo   = (const float*)d_in[8];

  size_t off = 0;
  auto carve = [&](size_t bytes) {
    void* p = (char*)d_ws + off;
    off += (bytes + 255) & ~(size_t)255;
    return p;
  };
  const size_t nX = (size_t)MROWS * DIMD;   // 8.4M elems
  const size_t nW = (size_t)DIMD * DIMD;    // 4.2M elems
  unsigned short* xb  = (unsigned short*)carve(nX * 2);
  unsigned short* wqb = (unsigned short*)carve(nW * 2);
  unsigned short* wkb = (unsigned short*)carve(nW * 2);
  unsigned short* wvb = (unsigned short*)carve(nW * 2);
  unsigned short* wob = (unsigned short*)carve(nW * 2);
  unsigned short* qb  = (unsigned short*)carve(nX * 2);
  unsigned short* kb  = (unsigned short*)carve(nX * 2);
  unsigned short* vb  = (unsigned short*)carve(nX * 2);
  unsigned short* ab  = (unsigned short*)carve(nX * 2);

  // 1) fp32 -> bf16
  cvt_f32_bf16<<<(int)((nX + 255) / 256), 256, 0, stream>>>(x,  xb,  (int)nX);
  cvt_f32_bf16<<<(int)((nW + 255) / 256), 256, 0, stream>>>(wq, wqb, (int)nW);
  cvt_f32_bf16<<<(int)((nW + 255) / 256), 256, 0, stream>>>(wk, wkb, (int)nW);
  cvt_f32_bf16<<<(int)((nW + 255) / 256), 256, 0, stream>>>(wv, wvb, (int)nW);
  cvt_f32_bf16<<<(int)((nW + 255) / 256), 256, 0, stream>>>(wo, wob, (int)nW);

  // 2) Q/K/V projections: [4096,2048] @ [2048,2048]^T (bf16 out)
  dim3 ggrid(MROWS / 128, DIMD / 128);
  gemm_bt<0><<<ggrid, 256, 0, stream>>>(xb, wqb, qb, MROWS, DIMD, DIMD);
  gemm_bt<0><<<ggrid, 256, 0, stream>>>(xb, wkb, kb, MROWS, DIMD, DIMD);
  gemm_bt<0><<<ggrid, 256, 0, stream>>>(xb, wvb, vb, MROWS, DIMD, DIMD);

  // 3) RoPE on Q, K
  int pairs = BATCH * SEQL * NHEADS * (HEADDIM / 2);
  rope_qk<<<(pairs + 255) / 256, 256, 0, stream>>>(qb, kb, fcos, fsin, pairs);

  // 4) causal flash attention
  dim3 fgrid(SEQL / 128, NHEADS, BATCH);
  flash_attn<<<fgrid, 256, 0, stream>>>(qb, kb, vb, ab);

  // 5) output projection -> fp32 d_out
  gemm_bt<1><<<ggrid, 256, 0, stream>>>(ab, wob, (float*)d_out, MROWS, DIMD, DIMD);
}
